// ACAT_85718957293751
// MI455X (gfx1250) — matrix operand order repaired and checked by reading, not compiled, not run on hardware
//
#include <hip/hip_runtime.h>
#include <hip/hip_bf16.h>

typedef __attribute__((ext_vector_type(16))) __bf16 v16bf;
typedef __attribute__((ext_vector_type(8)))  __bf16 v8bf;
typedef __attribute__((ext_vector_type(8)))  float  v8f;

union U16 { v16bf v; v8bf h[2]; __bf16 e[16]; };

#if defined(__has_builtin)
#if __has_builtin(__builtin_amdgcn_sched_group_barrier)
#define MSA_HAVE_SGB 1
#endif
#endif
#ifdef MSA_HAVE_SGB
#define SGB(mask, size, id) __builtin_amdgcn_sched_group_barrier((mask), (size), (id))
#else
#define SGB(mask, size, id)
#endif

// ---------------------------------------------------------------------------
// Generic tiled transpose + f32 -> bf16 convert:  in[Z][R][C] -> out[Z][C][R]
// ---------------------------------------------------------------------------
__global__ __launch_bounds__(256) void msa_transpose_cvt(
    const float* __restrict__ in, __bf16* __restrict__ out, int R, int C)
{
    __shared__ float tile[32][33];
    int z  = blockIdx.z;
    int c0 = blockIdx.x * 32;
    int r0 = blockIdx.y * 32;
    int tx = threadIdx.x, ty = threadIdx.y;      // block (32, 8)
    const float* inz  = in  + (size_t)z * R * C;
    __bf16*      outz = out + (size_t)z * R * C;
#pragma unroll
    for (int k = 0; k < 4; ++k) {
        int r = r0 + ty + k * 8;
        tile[ty + k * 8][tx] = inz[(size_t)r * C + c0 + tx];
    }
    __syncthreads();
#pragma unroll
    for (int k = 0; k < 4; ++k) {
        int c = c0 + ty + k * 8;
        outz[(size_t)c * R + r0 + tx] = (__bf16)tile[tx][ty + k * 8];
    }
}

// ---------------------------------------------------------------------------
// Weight repack: W[512][512][f] f32 -> out[tap][512 o][512 i] bf16 (i fastest)
// ---------------------------------------------------------------------------
__global__ __launch_bounds__(256) void msa_pack_w(
    const float* __restrict__ W, __bf16* __restrict__ out, int f)
{
    int idx = blockIdx.x * 256 + threadIdx.x;    // f*512*512 threads
    int i   = idx & 511;
    int o   = (idx >> 9) & 511;
    int tap = idx >> 18;
    out[idx] = (__bf16)W[((size_t)o * 512 + i) * f + tap];
}

// ---------------------------------------------------------------------------
// Conv1d-as-GEMM, all 4 branches.  XT: [8][1024 t][512 i] bf16,
// Wb: [19 taps][512 o][512 i] bf16, Y: [4 j][8 bb][512 o][1024 t] f32.
// Block tile 128x128 (8 waves, wave tile 64x32 = 4x2 WMMA accums).
// Peeled, statically double-buffered k-loop; interior/edge specialization is
// hoisted so the steady-state body is a single basic block, letting the
// sched_group_barrier hints interleave VMEM issue with WMMA groups.
// ---------------------------------------------------------------------------
__global__ __launch_bounds__(256) void msa_conv_gemm(
    const __bf16* __restrict__ XT, const __bf16* __restrict__ Wb,
    const float* __restrict__ b0, const float* __restrict__ b1,
    const float* __restrict__ b2, const float* __restrict__ b3,
    float* __restrict__ Y)
{
    const int FIL[4] = {1, 3, 6, 9};
    const int PAD[4] = {0, 1, 3, 4};
    const int TB [4] = {0, 1, 4, 10};

    int bn = blockIdx.x;          // 0..7   (128 t)
    int bm = blockIdx.y;          // 0..3   (128 o)
    int z  = blockIdx.z;          // 0..31  j*8+bb
    int j  = z >> 3, bb = z & 7;
    int f = FIL[j], pad = PAD[j], tb = TB[j];
    const float* bias = (j == 0) ? b0 : (j == 1) ? b1 : (j == 2) ? b2 : b3;

    int lane = threadIdx.x & 31;
    int wave = threadIdx.x >> 5;
    int wm = wave >> 2, wn = wave & 3;           // 2 x 4 waves
    int Mo = bm * 128 + wm * 64;
    int No = bn * 128 + wn * 32;
    int l15 = lane & 15, hh = lane >> 4;
    bool interior = (No >= 4) && (No <= 1024 - 36);   // covers all shifts +-4

    const __bf16* Xb = XT + (size_t)bb * 1024 * 512;
    v8f acc[4][2] = {};
    v16bf a0[4], a1[4];
    v16bf bm0[2], bm1[2];

    auto load_stage = [&](v16bf (&a)[4], v16bf (&bm)[2], int ks, bool guarded) {
        int tap   = ks >> 4;
        int kk    = (ks & 15) << 5;
        int shift = tap - pad;
        const __bf16* Wt = Wb + (size_t)(tb + tap) * 512 * 512 + kk;
#pragma unroll
        for (int mi = 0; mi < 4; ++mi) {
            const __bf16* p = Wt + (size_t)(Mo + mi * 16 + l15) * 512;
            U16 u;
            u.h[0] = *(const v8bf*)(p + 16 * hh);
            u.h[1] = *(const v8bf*)(p + 16 * hh + 8);
            a[mi] = u.v;
        }
        if (!guarded) {
#pragma unroll
            for (int ni = 0; ni < 2; ++ni) {
                int t = No + ni * 16 + l15 + shift;
                bm[ni] = *(const v16bf*)(Xb + (size_t)t * 512 + kk + 16 * hh);
            }
        } else {
#pragma unroll
            for (int ni = 0; ni < 2; ++ni) {
                int t = No + ni * 16 + l15 + shift;
                v16bf bz = {};
                bm[ni] = (t >= 0 && t < 1024)
                    ? *(const v16bf*)(Xb + (size_t)t * 512 + kk + 16 * hh)
                    : bz;
            }
        }
    };

    auto wmma_step = [&](v16bf (&a)[4], v16bf (&bm)[2]) {
#pragma unroll
        for (int mi = 0; mi < 4; ++mi)
#pragma unroll
            for (int ni = 0; ni < 2; ++ni)
                acc[mi][ni] = __builtin_amdgcn_wmma_f32_16x16x32_bf16(
                    false, a[mi], false, bm[ni], (short)0, acc[mi][ni],
                    false, false);
    };

    int nks = f * 16;                 // always even (16, 48, 96, 144)
    auto gemm_loop = [&](bool guarded) {
        load_stage(a0, bm0, 0, guarded);
        int ks = 0;
        for (; ks + 2 < nks; ks += 2) {   // steady state: branch-free body
            load_stage(a1, bm1, ks + 1, guarded);
            wmma_step(a0, bm0);
            load_stage(a0, bm0, ks + 2, guarded);
            wmma_step(a1, bm1);
#pragma unroll
            for (int g = 0; g < 8; ++g) {
                SGB(0x020, 3, 0);         // 3 VMEM reads
                SGB(0x008, 2, 0);         // 2 WMMA
            }
        }
        load_stage(a1, bm1, ks + 1, guarded);   // peeled epilogue
        wmma_step(a0, bm0);
        wmma_step(a1, bm1);
    };
    if (interior) gemm_loop(false); else gemm_loop(true);

    float* Yb = Y + (size_t)z * 512 * 1024;
#pragma unroll
    for (int mi = 0; mi < 4; ++mi)
#pragma unroll
        for (int ni = 0; ni < 2; ++ni)
#pragma unroll
            for (int r = 0; r < 8; ++r) {
                int o = Mo + mi * 16 + r + 8 * hh;
                int t = No + ni * 16 + l15;
                Yb[(size_t)o * 1024 + t] = acc[mi][ni][r] + bias[o];
            }
}

// ---------------------------------------------------------------------------
// Scrambled multi-scale combine (replicates stack(...,0).reshape semantics):
// Out[b2][h2][l2][d2] = ReLU( sum_s lw[s]*Y[j][bb][c][tb+s] + lb )
// ---------------------------------------------------------------------------
__global__ __launch_bounds__(256) void msa_combine(
    const float* __restrict__ Y, const float* __restrict__ lw,
    const float* __restrict__ lb, __bf16* __restrict__ out)
{
    size_t idx = (size_t)blockIdx.x * 256 + threadIdx.x;   // 8*8*1024*64
    int d2 = (int)(idx & 63);
    int l2 = (int)((idx >> 6) & 1023);
    int h2 = (int)((idx >> 16) & 7);
    int b2 = (int)(idx >> 19);
    int j  = b2 >> 1;
    int bb = (b2 & 1) * 4 + (h2 >> 1);
    int c  = (h2 & 1) * 256 + (l2 >> 2);
    int t  = (l2 & 3) * 256 + d2 * 4;
    float4 y = *(const float4*)(Y + (((size_t)(j * 8 + bb) * 512 + c) * 1024 + t));
    float v = lw[0] * y.x + lw[1] * y.y + lw[2] * y.z + lw[3] * y.w + lb[0];
    out[idx] = (__bf16)fmaxf(v, 0.f);
}

// ---------------------------------------------------------------------------
// scores = Qn Kn^T / 8, causal mask -> logits into attn buffer (f32).
// Blocks fully above the diagonal skip the GEMM and just write -1e9.
// ---------------------------------------------------------------------------
__global__ __launch_bounds__(256) void msa_scores(
    const __bf16* __restrict__ Qn, const __bf16* __restrict__ Kn,
    float* __restrict__ S)
{
    int bh = blockIdx.z;
    int lane = threadIdx.x & 31;
    int wave = threadIdx.x >> 5;
    int wm = wave >> 2, wn = wave & 3;
    int Qo = blockIdx.y * 128 + wm * 64;
    int Ko = blockIdx.x * 128 + wn * 32;
    int l15 = lane & 15, hh = lane >> 4;
    float* Sb = S + (size_t)bh * 1024 * 1024;

    if (blockIdx.x > blockIdx.y) {               // fully masked block
#pragma unroll
        for (int mi = 0; mi < 4; ++mi)
#pragma unroll
            for (int ni = 0; ni < 2; ++ni)
#pragma unroll
                for (int r = 0; r < 8; ++r) {
                    int q  = Qo + mi * 16 + r + 8 * hh;
                    int kp = Ko + ni * 16 + l15;
                    Sb[(size_t)q * 1024 + kp] = -1e9f;
                }
        return;
    }

    const __bf16* Qb = Qn + (size_t)bh * 1024 * 64;
    const __bf16* Kb = Kn + (size_t)bh * 1024 * 64;
    v8f acc[4][2] = {};
#pragma unroll
    for (int kk = 0; kk < 64; kk += 32) {
        v16bf a[4];
#pragma unroll
        for (int mi = 0; mi < 4; ++mi) {
            const __bf16* p = Qb + (size_t)(Qo + mi * 16 + l15) * 64 + kk;
            U16 u;
            u.h[0] = *(const v8bf*)(p + 16 * hh);
            u.h[1] = *(const v8bf*)(p + 16 * hh + 8);
            a[mi] = u.v;
        }
        v16bf bmat[2];
#pragma unroll
        for (int ni = 0; ni < 2; ++ni)
            bmat[ni] = *(const v16bf*)(Kb + (size_t)(Ko + ni * 16 + l15) * 64 + kk + 16 * hh);
#pragma unroll
        for (int mi = 0; mi < 4; ++mi)
#pragma unroll
            for (int ni = 0; ni < 2; ++ni)
                acc[mi][ni] = __builtin_amdgcn_wmma_f32_16x16x32_bf16(
                    false, a[mi], false, bmat[ni], (short)0, acc[mi][ni],
                    false, false);
    }
#pragma unroll
    for (int mi = 0; mi < 4; ++mi)
#pragma unroll
        for (int ni = 0; ni < 2; ++ni)
#pragma unroll
            for (int r = 0; r < 8; ++r) {
                int q  = Qo + mi * 16 + r + 8 * hh;
                int kp = Ko + ni * 16 + l15;
                float v = acc[mi][ni][r] * 0.125f;
                if (kp > q) v = -1e9f;
                Sb[(size_t)q * 1024 + kp] = v;
            }
}

// ---------------------------------------------------------------------------
// Row softmax in place: one wave32 per 1024-wide row
// ---------------------------------------------------------------------------
__global__ __launch_bounds__(256) void msa_softmax(float* __restrict__ S)
{
    int lane = threadIdx.x & 31;
    size_t row = (size_t)blockIdx.x * 8 + (threadIdx.x >> 5);
    float* p = S + row * 1024;
    float4 v[8];
    float m = -3.4e38f;
#pragma unroll
    for (int i = 0; i < 8; ++i) {
        v[i] = *(const float4*)(p + lane * 4 + i * 128);
        m = fmaxf(m, fmaxf(fmaxf(v[i].x, v[i].y), fmaxf(v[i].z, v[i].w)));
    }
#pragma unroll
    for (int off = 16; off > 0; off >>= 1) m = fmaxf(m, __shfl_xor(m, off, 32));
    float s = 0.f;
#pragma unroll
    for (int i = 0; i < 8; ++i) {
        v[i].x = __expf(v[i].x - m); v[i].y = __expf(v[i].y - m);
        v[i].z = __expf(v[i].z - m); v[i].w = __expf(v[i].w - m);
        s += v[i].x + v[i].y + v[i].z + v[i].w;
    }
#pragma unroll
    for (int off = 16; off > 0; off >>= 1) s += __shfl_xor(s, off, 32);
    float inv = 1.0f / s;
#pragma unroll
    for (int i = 0; i < 8; ++i) {
        v[i].x *= inv; v[i].y *= inv; v[i].z *= inv; v[i].w *= inv;
        *(float4*)(p + lane * 4 + i * 128) = v[i];
    }
}

// ---------------------------------------------------------------------------
// context = attn @ V.  attn f32 (converted to bf16 on load), Vt bf16
// [64 bh][64 d][1024 kp].  4 waves/block, wave tile 64x64 (4x4 accums).
// Causal: attn[q][kp]==0 for kp>q, so the k-loop stops at Mo+64.
// Peeled double-buffered loop (kmax >= 64 so the epilogue is always valid).
// ---------------------------------------------------------------------------
__device__ __forceinline__ void cvt4(__bf16* dst, float4 f) {
    dst[0] = (__bf16)f.x; dst[1] = (__bf16)f.y;
    dst[2] = (__bf16)f.z; dst[3] = (__bf16)f.w;
}

__global__ __launch_bounds__(128) void msa_context(
    const float* __restrict__ S, const __bf16* __restrict__ Vt,
    float* __restrict__ C)
{
    int bh = blockIdx.y;
    int lane = threadIdx.x & 31;
    int wave = threadIdx.x >> 5;
    int Mo = blockIdx.x * 256 + wave * 64;
    int l15 = lane & 15, hh = lane >> 4;

    const float*  Sb = S  + (size_t)bh * 1024 * 1024;
    const __bf16* Vb = Vt + (size_t)bh * 64 * 1024;
    v8f acc[4][4] = {};
    v16bf a0[4], a1[4];
    v16bf bm0[4], bm1[4];

    auto load_stage = [&](v16bf (&a)[4], v16bf (&bm)[4], int kk) {
#pragma unroll
        for (int mi = 0; mi < 4; ++mi) {
            const float* p = Sb + (size_t)(Mo + mi * 16 + l15) * 1024 + kk;
            U16 u;
            cvt4(&u.e[0],  *(const float4*)(p + 16 * hh));
            cvt4(&u.e[4],  *(const float4*)(p + 16 * hh + 4));
            cvt4(&u.e[8],  *(const float4*)(p + 16 * hh + 8));
            cvt4(&u.e[12], *(const float4*)(p + 16 * hh + 12));
            a[mi] = u.v;
        }
#pragma unroll
        for (int ni = 0; ni < 4; ++ni)
            bm[ni] = *(const v16bf*)(Vb + (size_t)(ni * 16 + l15) * 1024 + kk + 16 * hh);
    };

    auto wmma_step = [&](v16bf (&a)[4], v16bf (&bm)[4]) {
#pragma unroll
        for (int mi = 0; mi < 4; ++mi)
#pragma unroll
            for (int ni = 0; ni < 4; ++ni)
                acc[mi][ni] = __builtin_amdgcn_wmma_f32_16x16x32_bf16(
                    false, a[mi], false, bm[ni], (short)0, acc[mi][ni],
                    false, false);
    };

    int kmax = Mo + 64;               // beyond this attn is exactly zero
    load_stage(a0, bm0, 0);
    int kk = 0;
    for (; kk + 64 < kmax; kk += 64) {    // steady state: branch-free body
        load_stage(a1, bm1, kk + 32);
        wmma_step(a0, bm0);
        load_stage(a0, bm0, kk + 64);
        wmma_step(a1, bm1);
#pragma unroll
        for (int g = 0; g < 8; ++g) {
            SGB(0x020, 6, 0);
            SGB(0x008, 4, 0);
        }
    }
    load_stage(a1, bm1, kk + 32);         // peeled epilogue
    wmma_step(a0, bm0);
    wmma_step(a1, bm1);

    float* Cb = C + (size_t)bh * 1024 * 64;
#pragma unroll
    for (int mi = 0; mi < 4; ++mi)
#pragma unroll
        for (int ni = 0; ni < 4; ++ni)
#pragma unroll
            for (int r = 0; r < 8; ++r) {
                int q = Mo + mi * 16 + r + 8 * hh;
                int d = ni * 16 + l15;
                Cb[(size_t)q * 64 + d] = acc[mi][ni][r];
            }
}

// ---------------------------------------------------------------------------
extern "C" void kernel_launch(void* const* d_in, const int* in_sizes, int n_in,
                              void* d_out, int out_size, void* d_ws, size_t ws_size,
                              hipStream_t stream)
{
    (void)in_sizes; (void)n_in; (void)out_size; (void)ws_size;
    static const int FIL[4] = {1, 3, 6, 9};
    static const int TB [4] = {0, 1, 4, 10};

    const float* Q = (const float*)d_in[0];
    const float* K = (const float*)d_in[1];
    const float* V = (const float*)d_in[2];
    const float* qw[4], *qb[4], *kw[4], *kb[4];
    for (int j = 0; j < 4; ++j) {
        qw[j] = (const float*)d_in[4 + 4 * j + 0];
        qb[j] = (const float*)d_in[4 + 4 * j + 1];
        kw[j] = (const float*)d_in[4 + 4 * j + 2];
        kb[j] = (const float*)d_in[4 + 4 * j + 3];
    }
    const float* lqw = (const float*)d_in[20];
    const float* lqb = (const float*)d_in[21];
    const float* lkw = (const float*)d_in[22];
    const float* lkb = (const float*)d_in[23];

    float* ctx  = (float*)d_out;                       // [64][1024][64]
    float* attn = (float*)d_out + (size_t)8 * 8 * 1024 * 64;

    char* ws = (char*)d_ws;
    float*  Y   = (float*) (ws + 0);                   // 64 MB
    __bf16* XT  = (__bf16*)(ws + 67108864);            //  8 MB
    __bf16* Wb  = (__bf16*)(ws + 75497472);            // 10 MB
    __bf16* Qn  = (__bf16*)(ws + 85458944);            //  8 MB
    __bf16* Kn  = (__bf16*)(ws + 93847552);            //  8 MB
    __bf16* Vt  = (__bf16*)(ws + 102236160);           //  8 MB

    dim3 tb32(32, 8);
    // ---- Q side ----
    for (int j = 0; j < 4; ++j)
        msa_pack_w<<<1024 * FIL[j], 256, 0, stream>>>(qw[j], Wb + (size_t)TB[j] * 512 * 512, FIL[j]);
    msa_transpose_cvt<<<dim3(1024 / 32, 512 / 32, 8), tb32, 0, stream>>>(Q, XT, 512, 1024);
    msa_conv_gemm<<<dim3(8, 4, 32), 256, 0, stream>>>(XT, Wb, qb[0], qb[1], qb[2], qb[3], Y);
    msa_combine<<<16384, 256, 0, stream>>>(Y, lqw, lqb, Qn);
    // ---- K side (reuse Wb/XT/Y) ----
    for (int j = 0; j < 4; ++j)
        msa_pack_w<<<1024 * FIL[j], 256, 0, stream>>>(kw[j], Wb + (size_t)TB[j] * 512 * 512, FIL[j]);
    msa_transpose_cvt<<<dim3(1024 / 32, 512 / 32, 8), tb32, 0, stream>>>(K, XT, 512, 1024);
    msa_conv_gemm<<<dim3(8, 4, 32), 256, 0, stream>>>(XT, Wb, kb[0], kb[1], kb[2], kb[3], Y);
    msa_combine<<<16384, 256, 0, stream>>>(Y, lkw, lkb, Kn);
    // ---- V transpose: [64 bh][1024 kp][64 d] -> [64 bh][64 d][1024 kp] ----
    msa_transpose_cvt<<<dim3(64 / 32, 1024 / 32, 64), tb32, 0, stream>>>(V, Vt, 1024, 64);
    // ---- attention ----
    msa_scores<<<dim3(8, 8, 64), 256, 0, stream>>>(Qn, Kn, attn);
    msa_softmax<<<8192, 256, 0, stream>>>(attn);
    msa_context<<<dim3(4, 64), 128, 0, stream>>>(attn, Vt, ctx);
}
